// OldMultiheadAttention_64269890617995
// MI455X (gfx1250) — matrix operand order repaired and checked by reading, not compiled, not run on hardware
//
#include <hip/hip_runtime.h>
#include <hip/hip_bf16.h>

#define MODEL_DIM 1024
#define HEAD_DIM  64
#define N_HEAD    16
#define B_SZ      4
#define SEQ       2048
#define EPS       1e-5f

typedef __attribute__((ext_vector_type(16))) __bf16 v16bf;
typedef __attribute__((ext_vector_type(8)))  float  v8f;
typedef __attribute__((ext_vector_type(4)))  unsigned int u32x4;
typedef __attribute__((ext_vector_type(8)))  int i32x8;
typedef __attribute__((ext_vector_type(4)))  int i32x4;

#if defined(__has_builtin)
#if __has_builtin(__builtin_amdgcn_tensor_load_to_lds)
#define HAVE_TDM 1
#endif
#endif
#ifndef HAVE_TDM
#define HAVE_TDM 0
#endif

union Frag16 {
  uint4 u4[2];
  v16bf v;
};

__device__ __forceinline__ v8f wmma_bf16(v16bf a, v16bf b, v8f c) {
  return __builtin_amdgcn_wmma_f32_16x16x32_bf16(
      false, a, false, b, (short)0, c, false, false);
}

// Low 32 bits of a generic pointer to __shared__ == LDS byte address
// (LDS aperture keeps offset in addr[31:0], ISA sec 10.2).
__device__ __forceinline__ unsigned lds_byte_of(const void* p) {
  return (unsigned)(unsigned long long)p;
}

#if HAVE_TDM
// Tensor Data Mover: load a 2D bf16 tile [tile_d1 rows x tile_d0 elems]
// (row stride = stride_elems) from global to contiguous LDS.
__device__ __forceinline__ void tdm_load_2d(unsigned lds_byte, const void* gptr,
                                            unsigned tile_d0, unsigned tile_d1,
                                            unsigned stride_elems) {
  unsigned long long ga = (unsigned long long)gptr;
  unsigned td0 = stride_elems;   // tensor dim0 (OOB bound; tile never exceeds it)
  unsigned td1 = 1u << 20;       // tensor dim1: large, no OOB in our launches
  u32x4 g0;
  g0[0] = 1u;                                            // count=1, user mode
  g0[1] = lds_byte;                                      // lds_addr
  g0[2] = (unsigned)ga;                                  // global_addr[31:0]
  g0[3] = ((unsigned)(ga >> 32) & 0x01FFFFFFu)           // global_addr[56:32]
        | 0x80000000u;                                   // type=2 ("image")
  i32x8 g1;
  g1[0] = 0x00010000;                                    // data_size=1 (2 bytes)
  g1[1] = (int)((td0 & 0xFFFFu) << 16);                  // tensor_dim0 lo16
  g1[2] = (int)(((td0 >> 16) & 0xFFFFu) | ((td1 & 0xFFFFu) << 16));
  g1[3] = (int)(((td1 >> 16) & 0xFFFFu) | ((tile_d0 & 0xFFFFu) << 16));
  g1[4] = (int)(tile_d1 & 0xFFFFu);                      // tile_dim1 (tile_dim2=0)
  g1[5] = (int)stride_elems;                             // tensor_dim0_stride lo32
  g1[6] = 0;
  g1[7] = 0;
  i32x4 gz; gz[0] = gz[1] = gz[2] = gz[3] = 0;
#if __clang_major__ >= 23
  i32x8 gz8; gz8[0]=gz8[1]=gz8[2]=gz8[3]=gz8[4]=gz8[5]=gz8[6]=gz8[7]=0;
  __builtin_amdgcn_tensor_load_to_lds(g0, g1, gz, gz, gz8, 0);
#else
  __builtin_amdgcn_tensor_load_to_lds(g0, g1, gz, gz, 0);
#endif
}
#endif

// ---------------------------------------------------------------------------
// K0: weight convert + transpose: W f32 [K][N] -> Wt bf16 [N][K]
// ---------------------------------------------------------------------------
__global__ __launch_bounds__(256) void wconv_kernel(const float* __restrict__ W,
                                                    __bf16* __restrict__ Wt) {
  int idx = blockIdx.x * 256 + threadIdx.x;
  int k = idx >> 10;
  int n = idx & 1023;
  Wt[(size_t)n * MODEL_DIM + k] = (__bf16)W[idx];
}

// ---------------------------------------------------------------------------
// K1/K3: GEMM C[M,1024] = A[M,1024] * Bt^T (+bias [+residual])
// Block tile 128x128, 8 waves, wave tile 32x64 -> 8 WMMA per k-step.
// B tile (and A tile when bf16) staged by the Tensor Data Mover.
// ---------------------------------------------------------------------------
template <bool A_BF16, bool FUSE_RES>
__global__ __launch_bounds__(256) void gemm_kernel(const void* __restrict__ Ap,
                                                   const __bf16* __restrict__ Bt,
                                                   const float* __restrict__ bias,
                                                   const float* __restrict__ residual,
                                                   void* __restrict__ Cp,
                                                   float scale) {
  __shared__ __bf16 lds_a[128 * 32];
  __shared__ __bf16 lds_b[128 * 32];

  const int tid  = threadIdx.x;
  const int lane = tid & 31;
  const int w    = tid >> 5;
  const int wm   = (w & 3) * 32;
  const int wn   = (w >> 2) * 64;
  const int m0   = blockIdx.y * 128;
  const int n0   = blockIdx.x * 128;
  const int ln   = lane & 15;
  const int lh   = (lane < 16) ? 0 : 8;    // A-frag K base
  const int lk   = (lane < 16) ? 0 : 16;   // unused; B-frag K base is lh too

  v8f acc[2][4];
  v8f zero = {};
#pragma unroll
  for (int i = 0; i < 2; ++i)
#pragma unroll
    for (int j = 0; j < 4; ++j) acc[i][j] = zero;

  for (int kk = 0; kk < MODEL_DIM; kk += 32) {
    __syncthreads();
#if HAVE_TDM
    if (tid < 32) {
      tdm_load_2d(lds_byte_of(lds_b), Bt + (size_t)n0 * MODEL_DIM + kk,
                  32, 128, MODEL_DIM);
      if (A_BF16)
        tdm_load_2d(lds_byte_of(lds_a),
                    (const __bf16*)Ap + (size_t)m0 * MODEL_DIM + kk,
                    32, 128, MODEL_DIM);
    }
#else
#pragma unroll
    for (int i = 0; i < 2; ++i) {
      int idx = tid + i * 256;
      int r = idx >> 2, c8 = (idx & 3) * 8;
      *(uint4*)&lds_b[r * 32 + c8] =
          *(const uint4*)(Bt + (size_t)(n0 + r) * MODEL_DIM + kk + c8);
      if (A_BF16)
        *(uint4*)&lds_a[r * 32 + c8] =
            *(const uint4*)((const __bf16*)Ap + (size_t)(m0 + r) * MODEL_DIM + kk + c8);
    }
#endif
    if (!A_BF16) {
      const float* A = (const float*)Ap;
#pragma unroll
      for (int i = 0; i < 4; ++i) {
        int idx = tid + i * 256;
        int r = idx >> 3, c4 = (idx & 7) * 4;
        const float* src = A + (size_t)(m0 + r) * MODEL_DIM + kk + c4;
        float4 f = *(const float4*)src;
        __bf16 t4[4] = {(__bf16)f.x, (__bf16)f.y, (__bf16)f.z, (__bf16)f.w};
        *(uint2*)&lds_a[r * 32 + c4] = *(uint2*)t4;
        if (kk + 32 < MODEL_DIM) __builtin_prefetch(src + 32, 0, 0);
      }
    } else if (kk + 32 < MODEL_DIM) {
      // warm GL2 for next tiles (global_prefetch_b8)
      int r = tid & 127;
      const __bf16* nxt = (tid < 128)
          ? Bt + (size_t)(n0 + r) * MODEL_DIM + kk + 32
          : (const __bf16*)Ap + (size_t)(m0 + r) * MODEL_DIM + kk + 32;
      __builtin_prefetch(nxt, 0, 0);
    }
#if HAVE_TDM
    if (tid < 32) __builtin_amdgcn_s_wait_tensorcnt(0);
#endif
    __syncthreads();

    Frag16 fa[2], fb[4];
#pragma unroll
    for (int i = 0; i < 2; ++i) {
      const __bf16* pa = &lds_a[(wm + i * 16 + ln) * 32 + lh];
      fa[i].u4[0] = *(const uint4*)pa;
      fa[i].u4[1] = *(const uint4*)(pa + 16);
    }
#pragma unroll
    for (int j = 0; j < 4; ++j) {
      const __bf16* pb = &lds_b[(wn + j * 16 + ln) * 32 + lh];
      fb[j].u4[0] = *(const uint4*)pb;
      fb[j].u4[1] = *(const uint4*)(pb + 16);
    }
#pragma unroll
    for (int i = 0; i < 2; ++i)
#pragma unroll
      for (int j = 0; j < 4; ++j)
        acc[i][j] = wmma_bf16(fa[i].v, fb[j].v, acc[i][j]);
  }

  const int r8 = (lane >= 16) ? 8 : 0;
#pragma unroll
  for (int i = 0; i < 2; ++i) {
    int rbase = m0 + wm + i * 16 + r8;
#pragma unroll
    for (int j = 0; j < 4; ++j) {
      int col = n0 + wn + j * 16 + ln;
      float bv = bias[col];
#pragma unroll
      for (int e = 0; e < 8; ++e) {
        size_t o = (size_t)(rbase + e) * MODEL_DIM + col;
        if (FUSE_RES) {
          ((float*)Cp)[o] = acc[i][j][e] + bv + residual[o];
        } else {
          ((__bf16*)Cp)[o] = (__bf16)((acc[i][j][e] + bv) * scale);
        }
      }
    }
  }
}

// ---------------------------------------------------------------------------
// K2: flash attention.  Block = (b, h, 256 q-rows); wave owns 32 q-rows.
// K tile via TDM; V transposed manually; 32 WMMA per 64-key block per wave.
// ---------------------------------------------------------------------------
__global__ __launch_bounds__(256) void attn_kernel(const __bf16* __restrict__ Qb,
                                                   const __bf16* __restrict__ Kb,
                                                   const __bf16* __restrict__ Vb,
                                                   __bf16* __restrict__ Ob) {
  __shared__ __bf16 k_lds[64 * 64];        // [key][d]
  __shared__ __bf16 vT_lds[64 * 64];       // [d][key]
  __shared__ __bf16 p_lds[8 * 32 * 64];    // per-wave [row][key]

  const int tid  = threadIdx.x;
  const int lane = tid & 31;
  const int w    = tid >> 5;
  const int ln   = lane & 15;
  const int lh   = (lane < 16) ? 0 : 8;
  const int lk   = (lane < 16) ? 0 : 16;
  const int prow = (lane >= 16) ? 8 : 0;
  const int b    = blockIdx.z;
  const int h    = blockIdx.y;
  const int q0   = blockIdx.x * 256;

  // Q fragments: 2 row tiles x 2 k-steps, kept in registers for the whole pass
  Frag16 aq[2][2];
#pragma unroll
  for (int rt = 0; rt < 2; ++rt) {
    const __bf16* qp =
        Qb + (size_t)(b * SEQ + q0 + w * 32 + rt * 16 + ln) * MODEL_DIM + h * HEAD_DIM;
#pragma unroll
    for (int kd = 0; kd < 2; ++kd) {
      const __bf16* p = qp + kd * 32 + lh;
      aq[rt][kd].u4[0] = *(const uint4*)p;
      aq[rt][kd].u4[1] = *(const uint4*)(p + 16);
    }
  }

  float mr[2][8], lr[2][8];
  v8f oa[2][4];
  v8f zero = {};
#pragma unroll
  for (int rt = 0; rt < 2; ++rt) {
#pragma unroll
    for (int e = 0; e < 8; ++e) { mr[rt][e] = -1e30f; lr[rt][e] = 0.f; }
#pragma unroll
    for (int t = 0; t < 4; ++t) oa[rt][t] = zero;
  }

  for (int kb = 0; kb < SEQ; kb += 64) {
    __syncthreads();
#if HAVE_TDM
    if (tid < 32)
      tdm_load_2d(lds_byte_of(k_lds),
                  Kb + (size_t)(b * SEQ + kb) * MODEL_DIM + h * HEAD_DIM,
                  64, 64, MODEL_DIM);
#endif
#pragma unroll
    for (int i = 0; i < 2; ++i) {
      int idx = tid + i * 256;
      int r = idx >> 3, c8 = (idx & 7) * 8;
      size_t goff = (size_t)(b * SEQ + kb + r) * MODEL_DIM + h * HEAD_DIM + c8;
#if !HAVE_TDM
      *(uint4*)&k_lds[r * 64 + c8] = *(const uint4*)(Kb + goff);
#endif
      uint4 vv = *(const uint4*)(Vb + goff);
      __bf16 tmp[8];
      *(uint4*)tmp = vv;
#pragma unroll
      for (int j = 0; j < 8; ++j) vT_lds[(c8 + j) * 64 + r] = tmp[j];
      if (kb + 64 < SEQ) {
        __builtin_prefetch(Kb + goff + 64 * MODEL_DIM, 0, 0);
        __builtin_prefetch(Vb + goff + 64 * MODEL_DIM, 0, 0);
      }
    }
#if HAVE_TDM
    if (tid < 32) __builtin_amdgcn_s_wait_tensorcnt(0);
#endif
    __syncthreads();

    // scores S = Q * K^T : K-fragments shared by both row tiles
    v8f s[2][4];
#pragma unroll
    for (int t = 0; t < 4; ++t) {
      Frag16 fk[2];
#pragma unroll
      for (int kd = 0; kd < 2; ++kd) {
        const __bf16* p = &k_lds[(t * 16 + ln) * 64 + kd * 32 + lh];
        fk[kd].u4[0] = *(const uint4*)p;
        fk[kd].u4[1] = *(const uint4*)(p + 16);
      }
#pragma unroll
      for (int rt = 0; rt < 2; ++rt) {
        v8f z = zero;
        z = wmma_bf16(aq[rt][0].v, fk[0].v, z);
        z = wmma_bf16(aq[rt][1].v, fk[1].v, z);
        s[rt][t] = z;
      }
    }

    // online softmax
#pragma unroll
    for (int rt = 0; rt < 2; ++rt) {
      float nm[8], al[8], rs[8];
#pragma unroll
      for (int e = 0; e < 8; ++e) {
        float mx = fmaxf(fmaxf(s[rt][0][e], s[rt][1][e]),
                         fmaxf(s[rt][2][e], s[rt][3][e]));
        mx = fmaxf(mx, __shfl_xor(mx, 1, 32));
        mx = fmaxf(mx, __shfl_xor(mx, 2, 32));
        mx = fmaxf(mx, __shfl_xor(mx, 4, 32));
        mx = fmaxf(mx, __shfl_xor(mx, 8, 32));
        nm[e] = fmaxf(mr[rt][e], mx);
        al[e] = __expf(mr[rt][e] - nm[e]);
        mr[rt][e] = nm[e];
        rs[e] = 0.f;
      }
#pragma unroll
      for (int t = 0; t < 4; ++t)
#pragma unroll
        for (int e = 0; e < 8; ++e) {
          float p = __expf(s[rt][t][e] - nm[e]);
          s[rt][t][e] = p;
          rs[e] += p;
        }
#pragma unroll
      for (int e = 0; e < 8; ++e) {
        float r = rs[e];
        r += __shfl_xor(r, 1, 32);
        r += __shfl_xor(r, 2, 32);
        r += __shfl_xor(r, 4, 32);
        r += __shfl_xor(r, 8, 32);
        lr[rt][e] = lr[rt][e] * al[e] + r;
#pragma unroll
        for (int t = 0; t < 4; ++t) oa[rt][t][e] *= al[e];
      }
      // P: C-layout -> A-layout via per-wave LDS (same-wave DS is in-order)
#pragma unroll
      for (int t = 0; t < 4; ++t)
#pragma unroll
        for (int e = 0; e < 8; ++e)
          p_lds[w * 2048 + (rt * 16 + e + prow) * 64 + t * 16 + ln] =
              (__bf16)s[rt][t][e];
    }

    Frag16 ap[2][2];
#pragma unroll
    for (int rt = 0; rt < 2; ++rt)
#pragma unroll
      for (int kd = 0; kd < 2; ++kd) {
        const __bf16* p = &p_lds[w * 2048 + (rt * 16 + ln) * 64 + kd * 32 + lh];
        ap[rt][kd].u4[0] = *(const uint4*)p;
        ap[rt][kd].u4[1] = *(const uint4*)(p + 16);
      }

    // O += P * V : V-fragments shared by both row tiles
#pragma unroll
    for (int t = 0; t < 4; ++t) {
      Frag16 fv[2];
#pragma unroll
      for (int kd = 0; kd < 2; ++kd) {
        const __bf16* p = &vT_lds[(t * 16 + ln) * 64 + kd * 32 + lh];
        fv[kd].u4[0] = *(const uint4*)p;
        fv[kd].u4[1] = *(const uint4*)(p + 16);
      }
#pragma unroll
      for (int rt = 0; rt < 2; ++rt) {
        oa[rt][t] = wmma_bf16(ap[rt][0].v, fv[0].v, oa[rt][t]);
        oa[rt][t] = wmma_bf16(ap[rt][1].v, fv[1].v, oa[rt][t]);
      }
    }
  }

  // epilogue: O /= l, write bf16 [B,S,H*D]
#pragma unroll
  for (int rt = 0; rt < 2; ++rt) {
    int rbase = q0 + w * 32 + rt * 16 + prow;
#pragma unroll
    for (int e = 0; e < 8; ++e) {
      float inv = 1.f / lr[rt][e];
#pragma unroll
      for (int t = 0; t < 4; ++t) {
        int col = h * HEAD_DIM + t * 16 + ln;
        Ob[(size_t)(b * SEQ + rbase + e) * MODEL_DIM + col] =
            (__bf16)(oa[rt][t][e] * inv);
      }
    }
  }
}

// ---------------------------------------------------------------------------
// K4: in-place LayerNorm over rows of 1024 (one block per row)
// ---------------------------------------------------------------------------
__global__ __launch_bounds__(256) void ln_kernel(float* __restrict__ out,
                                                 const float* __restrict__ gamma,
                                                 const float* __restrict__ beta) {
  __shared__ float s1[256], s2[256];
  const int tid = threadIdx.x;
  float* row = out + (size_t)blockIdx.x * MODEL_DIM;
  float4 x = *(const float4*)(row + tid * 4);
  s1[tid] = x.x + x.y + x.z + x.w;
  s2[tid] = x.x * x.x + x.y * x.y + x.z * x.z + x.w * x.w;
  __syncthreads();
  for (int o = 128; o > 0; o >>= 1) {
    if (tid < o) { s1[tid] += s1[tid + o]; s2[tid] += s2[tid + o]; }
    __syncthreads();
  }
  float mu  = s1[0] * (1.f / MODEL_DIM);
  float var = s2[0] * (1.f / MODEL_DIM) - mu * mu;
  float inv = rsqrtf(var + EPS);
  float4 g = *(const float4*)(gamma + tid * 4);
  float4 bb = *(const float4*)(beta + tid * 4);
  float4 y;
  y.x = (x.x - mu) * inv * g.x + bb.x;
  y.y = (x.y - mu) * inv * g.y + bb.y;
  y.z = (x.z - mu) * inv * g.z + bb.z;
  y.w = (x.w - mu) * inv * g.w + bb.w;
  *(float4*)(row + tid * 4) = y;
}

// ---------------------------------------------------------------------------
extern "C" void kernel_launch(void* const* d_in, const int* in_sizes, int n_in,
                              void* d_out, int out_size, void* d_ws, size_t ws_size,
                              hipStream_t stream) {
  const float* query = (const float*)d_in[0];
  const float* key_  = (const float*)d_in[1];
  const float* value = (const float*)d_in[2];
  const float* Wq    = (const float*)d_in[3];
  const float* bq    = (const float*)d_in[4];
  const float* Wk    = (const float*)d_in[5];
  const float* bk    = (const float*)d_in[6];
  const float* Wv    = (const float*)d_in[7];
  const float* bv    = (const float*)d_in[8];
  const float* Wo    = (const float*)d_in[9];
  const float* bo    = (const float*)d_in[10];
  const float* gamma = (const float*)d_in[11];
  const float* beta  = (const float*)d_in[12];
  float* out = (float*)d_out;

  char* ws = (char*)d_ws;
  const size_t MB16 = (size_t)16 * 1024 * 1024;
  __bf16* qb  = (__bf16*)(ws);
  __bf16* kb  = (__bf16*)(ws + 1 * MB16);
  __bf16* vb  = (__bf16*)(ws + 2 * MB16);
  __bf16* ob  = (__bf16*)(ws + 3 * MB16);
  __bf16* WqT = (__bf16*)(ws + 4 * MB16);
  __bf16* WkT = WqT + 1024 * 1024;
  __bf16* WvT = WkT + 1024 * 1024;
  __bf16* WoT = WvT + 1024 * 1024;

  dim3 blk(256);

  wconv_kernel<<<4096, blk, 0, stream>>>(Wq, WqT);
  wconv_kernel<<<4096, blk, 0, stream>>>(Wk, WkT);
  wconv_kernel<<<4096, blk, 0, stream>>>(Wv, WvT);
  wconv_kernel<<<4096, blk, 0, stream>>>(Wo, WoT);

  dim3 ggrid(MODEL_DIM / 128, (B_SZ * SEQ) / 128);  // (8, 64)
  gemm_kernel<false, false><<<ggrid, blk, 0, stream>>>(query, WqT, bq, nullptr, qb, 0.125f);
  gemm_kernel<false, false><<<ggrid, blk, 0, stream>>>(key_,  WkT, bk, nullptr, kb, 1.0f);
  gemm_kernel<false, false><<<ggrid, blk, 0, stream>>>(value, WvT, bv, nullptr, vb, 1.0f);

  attn_kernel<<<dim3(SEQ / 256, N_HEAD, B_SZ), blk, 0, stream>>>(qb, kb, vb, ob);

  gemm_kernel<true, true><<<ggrid, blk, 0, stream>>>(ob, WoT, bo, query, out, 1.0f);

  ln_kernel<<<B_SZ * SEQ, blk, 0, stream>>>(out, gamma, beta);
}
